// FactorizationMachineModel_withGCN_29738353557516
// MI455X (gfx1250) — matrix operand order repaired and checked by reading, not compiled, not run on hardware
//
#include <hip/hip_runtime.h>

#define NN   16384      // N_NODES
#define KF   16384      // N_FEAT
#define DE   64         // EMBED
#define NB   4096       // BATCH
#define NE   524288     // N_EDGES
#define KT_TOT   (KF / 32)      // 512 K-tiles of 32
#define KCHUNKS  4
#define KT_PER   (KT_TOT / KCHUNKS)

typedef __attribute__((ext_vector_type(16))) __bf16 v16bf;
typedef __attribute__((ext_vector_type(8)))  float  v8f;
typedef __attribute__((ext_vector_type(4)))  float  f32x4;

union BfFrag { unsigned short u[16]; v16bf v; };

__device__ __forceinline__ unsigned short f2bf(float f) {
  unsigned int b = __float_as_uint(f);
  b += 0x7FFFu + ((b >> 16) & 1u);   // round-to-nearest-even
  return (unsigned short)(b >> 16);
}

__device__ __forceinline__ void atomF(float* p, float v) {
  __hip_atomic_fetch_add(p, v, __ATOMIC_RELAXED, __HIP_MEMORY_SCOPE_AGENT);
}

// ---- pack W [KF x DE] f32 row-major -> bf16 B-fragments (WMMA lane layout) ----
// fragment(kt,nt): lane L<16 holds col n=nt*16+L, K=kt*32+0..15 ; lane>=16: K=+16..31
__global__ void pack_w_kernel(const float* __restrict__ W, __bf16* __restrict__ wpack) {
  int wave = threadIdx.x >> 5;
  int lane = threadIdx.x & 31;
  int frag = blockIdx.x * 8 + wave;          // 2048 fragments total
  int kt = frag >> 2;
  int nt = frag & 3;
  int n  = nt * 16 + (lane & 15);
  int kb = kt * 32 + (lane >> 4) * 16;
  BfFrag f;
#pragma unroll
  for (int j = 0; j < 16; ++j)
    f.u[j] = f2bf(W[(size_t)(kb + j) * DE + n]);
  ((v16bf*)wpack)[(size_t)frag * 32 + lane] = f.v;
}

// ---- xw_partial[chunk] = X[:, Kchunk] @ W[Kchunk, :] : bf16 WMMA, f32 acc ----
// One wave = 16 rows x 64 cols over KF/KCHUNKS of the K dimension.
__global__ void gemm_kernel(const float* __restrict__ X, const __bf16* __restrict__ wpack,
                            float* __restrict__ xwp) {
  int wave  = threadIdx.x >> 5;
  int lane  = threadIdx.x & 31;
  int m0    = blockIdx.x * 128 + wave * 16;
  int chunk = blockIdx.y;
  int row   = m0 + (lane & 15);
  int khalf = (lane >> 4) * 16;              // lane<16 -> K base 0 ; lane>=16 -> +16
  const float* xr = X + (size_t)row * KF;
  const v16bf* wp = (const v16bf*)wpack;

  v8f acc[4] = {};

  int kt0 = chunk * KT_PER;
  for (int kt = kt0; kt < kt0 + KT_PER; ++kt) {
    int kk = kt * 32 + khalf;
    const f32x4* p = (const f32x4*)(xr + kk);       // K = kk .. kk+7
    const f32x4* q = (const f32x4*)(xr + kk + 8);   // K = kk+8 .. kk+15
    f32x4 a0 = __builtin_nontemporal_load(p);       // X streamed once: NT hint,
    f32x4 a1 = __builtin_nontemporal_load(p + 1);   // keep packed-W hot in L2
    f32x4 b0 = __builtin_nontemporal_load(q);
    f32x4 b1 = __builtin_nontemporal_load(q + 1);

    BfFrag af;
#pragma unroll
    for (int j = 0; j < 4; ++j) {
      af.u[j]      = f2bf(a0[j]);
      af.u[4 + j]  = f2bf(a1[j]);
      af.u[8 + j]  = f2bf(b0[j]);
      af.u[12 + j] = f2bf(b1[j]);
    }

#pragma unroll
    for (int nt = 0; nt < 4; ++nt) {
      v16bf bf = wp[(size_t)(kt * 4 + nt) * 32 + lane];
      acc[nt] = __builtin_amdgcn_wmma_f32_16x16x32_bf16(
          false, af.v, false, bf, (short)0, acc[nt], false, false);
    }
  }

  // D layout: VGPR g -> row m0 + g + (lane>=16 ? 8 : 0), col = lane&15 within tile
  float* outp = xwp + (size_t)chunk * NN * DE;
  int nb    = lane & 15;
  int rbase = m0 + (lane >> 4) * 8;
#pragma unroll
  for (int nt = 0; nt < 4; ++nt)
#pragma unroll
    for (int g = 0; g < 8; ++g)
      outp[(size_t)(rbase + g) * DE + nt * 16 + nb] = acc[nt][g];
}

// xw = sum of 4 K-chunk partials (deterministic split-K reduction)
__global__ void reduce_xw_kernel(const float* __restrict__ xwp, float* __restrict__ xw) {
  int i = blockIdx.x * blockDim.x + threadIdx.x;   // exact NN*DE threads
  xw[i] = (xwp[i] + xwp[i + (size_t)NN * DE]) +
          (xwp[i + 2 * (size_t)NN * DE] + xwp[i + 3 * (size_t)NN * DE]);
}

// ---- degree / normalization / aggregation ----
__global__ void init_deg_kernel(float* __restrict__ deg) {
  int i = blockIdx.x * blockDim.x + threadIdx.x;
  if (i < NN) deg[i] = 1.0f;                 // self loop
}

__global__ void deg_kernel(const int* __restrict__ dst, float* __restrict__ deg) {
  int e = blockIdx.x * blockDim.x + threadIdx.x;
  if (e < NE) atomF(&deg[dst[e]], 1.0f);
}

__global__ void dinv_kernel(const float* __restrict__ deg, float* __restrict__ dinv) {
  int i = blockIdx.x * blockDim.x + threadIdx.x;
  if (i < NN) dinv[i] = rsqrtf(deg[i]);
}

// xout[i,d] = xw[i,d] * dinv[i]^2   (self-loop term; also initializes xout)
__global__ void selfloop_kernel(const float* __restrict__ xw, const float* __restrict__ dinv,
                                float* __restrict__ xout) {
  int i = blockIdx.x * blockDim.x + threadIdx.x;   // exact NN*DE threads
  float dv = dinv[i >> 6];
  xout[i] = xw[i] * dv * dv;
}

// xout[dst,d] += xw[src,d] * dinv[src]*dinv[dst]
__global__ void scatter_kernel(const int* __restrict__ src, const int* __restrict__ dst,
                               const float* __restrict__ xw, const float* __restrict__ dinv,
                               float* __restrict__ xout) {
  long t = (long)blockIdx.x * blockDim.x + threadIdx.x;  // exact NE*64 threads
  int e = (int)(t >> 6);
  int d = (int)(t & 63);
  int s = src[e], q = dst[e];
  float norm = dinv[s] * dinv[q];
  atomF(&xout[(size_t)q * DE + d], xw[(size_t)s * DE + d] * norm);
}

// ---- FM head: out[b] = dot(emb[x0]+bg, emb[x1]+bg) + wlin[x0]+wlin[x1]+blin ----
__global__ void fm_kernel(const int* __restrict__ x, const float* __restrict__ xout,
                          const float* __restrict__ bg, const float* __restrict__ wlin,
                          const float* __restrict__ blin, float* __restrict__ out) {
  int b = blockIdx.x * blockDim.x + threadIdx.x;
  if (b >= NB) return;
  int i0 = x[2 * b], i1 = x[2 * b + 1];
  const float* e0 = xout + (size_t)i0 * DE;
  const float* e1 = xout + (size_t)i1 * DE;
  float fm = 0.0f;
#pragma unroll
  for (int d = 0; d < DE; ++d)
    fm += (e0[d] + bg[d]) * (e1[d] + bg[d]);
  out[b] = wlin[i0] + wlin[i1] + blin[0] + fm;
}

extern "C" void kernel_launch(void* const* d_in, const int* in_sizes, int n_in,
                              void* d_out, int out_size, void* d_ws, size_t ws_size,
                              hipStream_t stream) {
  const int*   x    = (const int*)d_in[0];
  const float* X    = (const float*)d_in[1];
  const int*   ei   = (const int*)d_in[2];
  const float* W    = (const float*)d_in[3];
  const float* bg   = (const float*)d_in[4];
  const float* wlin = (const float*)d_in[5];
  const float* blin = (const float*)d_in[6];
  float* out = (float*)d_out;

  char* ws = (char*)d_ws;
  __bf16* wpack = (__bf16*)ws;                          //  0MB: 2 MB packed W (bf16)
  float*  xwp   = (float*)(ws + (2u  << 20));           //  2MB: 16 MB split-K partials
  float*  xw    = (float*)(ws + (18u << 20));           // 18MB: 4 MB reduced X@W
  float*  xout  = (float*)(ws + (22u << 20));           // 22MB: 4 MB aggregated emb
  float*  deg   = (float*)(ws + (26u << 20));           // 26MB: 64 KB
  float*  dinv  = (float*)(ws + (26u << 20) + (1u << 16));

  const int* src = ei;
  const int* dst = ei + NE;

  pack_w_kernel<<<256, 256, 0, stream>>>(W, wpack);                 // 2048 frags / 8 per blk
  gemm_kernel<<<dim3(NN / 128, KCHUNKS), 256, 0, stream>>>(X, wpack, xwp);
  reduce_xw_kernel<<<(NN * DE) / 256, 256, 0, stream>>>(xwp, xw);
  init_deg_kernel<<<NN / 256, 256, 0, stream>>>(deg);
  deg_kernel<<<NE / 256, 256, 0, stream>>>(dst, deg);
  dinv_kernel<<<NN / 256, 256, 0, stream>>>(deg, dinv);
  selfloop_kernel<<<(NN * DE) / 256, 256, 0, stream>>>(xw, dinv, xout);
  scatter_kernel<<<(long)NE * DE / 256, 256, 0, stream>>>(src, dst, xw, dinv, xout);
  fm_kernel<<<NB / 256, 256, 0, stream>>>(x, xout, bg, wlin, blin, out);
}
